// MultiHeadAttentionDefenitive_3401614098810
// MI455X (gfx1250) — matrix operand order repaired and checked by reading, not compiled, not run on hardware
//
#include <hip/hip_runtime.h>

// Problem constants (match reference)
#define Bsz  2
#define Tsz  2048
#define Dsz  1024
#define Hsz  16
#define HDsz 64
#define Mtot (Bsz*Tsz)   // 4096
#define KT   64          // K elements staged per iteration
#define LDST 72          // padded LDS row stride in elements (64 data + 8 pad)

typedef unsigned short u16;
typedef __attribute__((ext_vector_type(16))) __bf16       v16bf;
typedef __attribute__((ext_vector_type(8)))  float        v8f;
typedef __attribute__((ext_vector_type(4)))  float        v4f;
typedef __attribute__((ext_vector_type(2)))  unsigned int v2u;
typedef __attribute__((ext_vector_type(4)))  unsigned int v4u;
typedef __attribute__((ext_vector_type(8)))  unsigned int v8u;
typedef __attribute__((ext_vector_type(4)))  int          i4;
typedef __attribute__((ext_vector_type(8)))  int          i8;

#if defined(__has_builtin)
#if __has_builtin(__builtin_amdgcn_tensor_load_to_lds)
#define HAVE_TDM 1
#endif
#endif
#ifndef HAVE_TDM
#define HAVE_TDM 0
#endif

// ---- helpers ---------------------------------------------------------------

static __device__ __forceinline__ u16 bf16bits(float f) {
  unsigned u = __builtin_bit_cast(unsigned, f);
  u += 0x7FFFu + ((u >> 16) & 1u);            // round-to-nearest-even
  return (u16)(u >> 16);
}
static __device__ __forceinline__ unsigned pkbf(float a, float b) {
  return (unsigned)bf16bits(a) | ((unsigned)bf16bits(b) << 16);
}
static __device__ __forceinline__ v8f vzero() {
  v8f z = {0.f,0.f,0.f,0.f,0.f,0.f,0.f,0.f};
  return z;
}
static __device__ __forceinline__ v16bf make_frag(v4u lo, v4u hi) {
  v8u w; w.lo = lo; w.hi = hi;
  return __builtin_bit_cast(v16bf, w);
}
static __device__ __forceinline__ v8f wmma_bf16(v16bf a, v16bf b, v8f c) {
  // D(f32 16x16) = A(bf16 16x32) * B(bf16 32x16) + C
  return __builtin_amdgcn_wmma_f32_16x16x32_bf16(false, a, false, b,
                                                 (short)0, c, false, false);
}
// A-fragment (16x32): lane's 16 bf16 = two 16B chunks at element offs {8*hf, 16+8*hf}
static __device__ __forceinline__ v16bf fragA(const u16* rowp, int hf) {
  v4u lo = *(const v4u*)(rowp + 8*hf);
  v4u hi = *(const v4u*)(rowp + 16 + 8*hf);
  return make_frag(lo, hi);
}
// B-fragment (32x16): lane's 16 bf16 = two 16B chunks at element offs {8*hf, 16+8*hf}
static __device__ __forceinline__ v16bf fragB(const u16* rowp, int hf) {
  const u16* p = rowp + 8*hf;
  v4u lo = *(const v4u*)(p);
  v4u hi = *(const v4u*)(p + 16);
  return make_frag(lo, hi);
}

// ---- Tensor Data Mover: DMA a rows x cols bf16 tile into LDS ---------------
// Pads 16B after every 128B row -> LDS row stride = 72 elements = LDST.
// D# per CDNA5 ISA (group0 + group1, groups 2/3 zero).

#if HAVE_TDM
static __device__ __forceinline__ void tdm_load_tile(
    const u16* gsrc, unsigned lds_off, int rows, int cols,
    int row_stride, int tensor_rows)
{
  unsigned long long ga = (unsigned long long)(size_t)gsrc;
  v4u g0;
  g0[0] = 1u;                                         // count=1, user mode
  g0[1] = lds_off;                                    // LDS byte address
  g0[2] = (unsigned)(ga & 0xFFFFFFFFu);               // global_addr[31:0]
  g0[3] = (unsigned)((ga >> 32) & 0x01FFFFFFu)        // global_addr[56:32]
        | 0x80000000u;                                // type = 2 ("image")
  i8 g1;
  // data_size=1 (2B), pad_enable, pad_interval=4 (32 DW = 128B), pad_amount=3 (4 DW = 16B)
  g1[0] = (1 << 16) | (1 << 20) | (4 << 22) | (3 << 25);
  g1[1] = (row_stride & 0xFFFF) << 16;                              // tensor_dim0 lo16
  g1[2] = ((row_stride >> 16) & 0xFFFF) | ((tensor_rows & 0xFFFF) << 16); // dim0 hi / dim1 lo
  g1[3] = ((tensor_rows >> 16) & 0xFFFF) | ((cols & 0xFFFF) << 16); // dim1 hi / tile_dim0
  g1[4] = rows & 0xFFFF;                                            // tile_dim1
  g1[5] = row_stride;                                               // tensor_dim0_stride lo32
  g1[6] = 0;
  g1[7] = 0;
  i4 gz4 = {0, 0, 0, 0};
  i8 gz8 = {0, 0, 0, 0, 0, 0, 0, 0};
  // 6-arg form (this toolchain): (g0, g1, g2, g3, g4, cpol)
  __builtin_amdgcn_tensor_load_to_lds(g0, g1, gz4, gz4, gz8, 0);
}
#endif

// Fallback: copy 128 x KT bf16 tile (row stride in elements) into LDS.
static __device__ __forceinline__ void stage_copy_bf16(u16* dst, const u16* src,
                                                       int src_stride, int tid) {
  const int row = tid >> 1;              // 0..127
  const int seg = (tid & 1) << 5;        // 0 or 32 elements (64B)
  const v4u* p = (const v4u*)(src + (size_t)row * src_stride + seg);
  v4u a = p[0], b = p[1], c = p[2], d = p[3];
  v4u* q = (v4u*)(dst + row * LDST + seg);
  q[0] = a; q[1] = b; q[2] = c; q[3] = d;
}

// ---- kernel 0: fp32 -> bf16 prepack ----------------------------------------

__global__ __launch_bounds__(256) void cvt_bf16(const float* __restrict__ src,
                                                u16* __restrict__ dst, int n4) {
  const int i = blockIdx.x * 256 + threadIdx.x;
  if (i < n4) {
    v4f a = ((const v4f*)src)[i];
    v2u o = { pkbf(a.x, a.y), pkbf(a.z, a.w) };
    ((v2u*)dst)[i] = o;
  }
}

// ---- kernel 1: QKV projections (y = x @ W^T), bf16 in / bf16 out -----------
// grid = (Mtot/128, Dsz/128, 3); mode 0->Q [B,H,T,HD], 1->K [B,H,T,HD], 2->V^T [B,H,HD,T]
// Double-buffered TDM staging: wave 0 issues tile i+1 before the block consumes
// tile i, then waits tensorcnt<=2 so only the newest two DMAs stay in flight.

__global__ __launch_bounds__(256) void qkv_wmma(
    const u16* __restrict__ xb,
    const u16* __restrict__ Wqb, const u16* __restrict__ Wkb, const u16* __restrict__ Wvb,
    u16* __restrict__ qws, u16* __restrict__ kws, u16* __restrict__ vtws)
{
  __shared__ __align__(16) u16 As[2][128 * LDST];
  __shared__ __align__(16) u16 Bs[2][128 * LDST];
  const int m0 = blockIdx.x * 128, n0 = blockIdx.y * 128, mode = blockIdx.z;
  const u16* W = (mode == 0) ? Wqb : (mode == 1) ? Wkb : Wvb;
  const int tid = threadIdx.x, lane = tid & 31, wid = tid >> 5;
  const int hf = lane >> 4, ln = lane & 15;
  const int wm = wid >> 1, wn = wid & 1;          // 4x2 wave grid: 32x64 per wave

  v8f acc[2][4];
  #pragma unroll
  for (int i = 0; i < 2; ++i)
    #pragma unroll
    for (int j = 0; j < 4; ++j) acc[i][j] = vzero();

#if HAVE_TDM
  if (wid == 0) {   // prologue: DMA tile 0 into buffer 0
    tdm_load_tile(xb + (size_t)m0 * Dsz, (unsigned)(size_t)(void*)As[0],
                  128, KT, Dsz, Mtot);
    tdm_load_tile(W  + (size_t)n0 * Dsz, (unsigned)(size_t)(void*)Bs[0],
                  128, KT, Dsz, Dsz);
  }
#endif

  for (int k0 = 0, it = 0; k0 < Dsz; k0 += KT, ++it) {
    const int cur = it & 1;
#if HAVE_TDM
    const int nxt = cur ^ 1;
    if (wid == 0) {
      if (k0 + KT < Dsz) {
        tdm_load_tile(xb + (size_t)m0 * Dsz + k0 + KT, (unsigned)(size_t)(void*)As[nxt],
                      128, KT, Dsz, Mtot);
        tdm_load_tile(W  + (size_t)n0 * Dsz + k0 + KT, (unsigned)(size_t)(void*)Bs[nxt],
                      128, KT, Dsz, Dsz);
        __builtin_amdgcn_s_wait_tensorcnt(2);   // current tile done; next in flight
      } else {
        __builtin_amdgcn_s_wait_tensorcnt(0);   // last tile: drain
      }
    }
#else
    stage_copy_bf16(As[cur], xb + (size_t)m0 * Dsz + k0, Dsz, tid);
    stage_copy_bf16(Bs[cur], W  + (size_t)n0 * Dsz + k0, Dsz, tid);
#endif
    __syncthreads();
    v16bf af[2][2];
    #pragma unroll
    for (int mf = 0; mf < 2; ++mf)
      #pragma unroll
      for (int kk = 0; kk < 2; ++kk)
        af[mf][kk] = fragA(As[cur] + (wm*32 + mf*16 + ln) * LDST + kk*32, hf);
    #pragma unroll
    for (int nf = 0; nf < 4; ++nf)
      #pragma unroll
      for (int kk = 0; kk < 2; ++kk) {
        v16bf bfr = fragB(Bs[cur] + (wn*64 + nf*16 + ln) * LDST + kk*32, hf);
        #pragma unroll
        for (int mf = 0; mf < 2; ++mf)
          acc[mf][nf] = wmma_bf16(af[mf][kk], bfr, acc[mf][nf]);
      }
    __syncthreads();
  }

  #pragma unroll
  for (int mf = 0; mf < 2; ++mf)
    #pragma unroll
    for (int nf = 0; nf < 4; ++nf)
      #pragma unroll
      for (int r = 0; r < 8; ++r) {
        const int m = m0 + wm*32 + mf*16 + r + 8*hf;   // token index in [0, B*T)
        const int n = n0 + wn*64 + nf*16 + ln;         // feature index in [0, D)
        const int b = m >> 11, tt = m & (Tsz - 1);
        const int h = n >> 6,  hd = n & (HDsz - 1);
        const u16 v = bf16bits(acc[mf][nf][r]);
        if (mode == 2)
          vtws[(((size_t)b*Hsz + h)*HDsz + hd)*Tsz + tt] = v;      // V^T
        else {
          u16* o = (mode == 0) ? qws : kws;
          o[(((size_t)b*Hsz + h)*Tsz + tt)*HDsz + hd] = v;         // Q, K
        }
      }
}

// ---- kernel 2: causal flash attention --------------------------------------
// grid = (Tsz/64, B*H), block = 128 (4 waves x 16 Q rows)

__global__ __launch_bounds__(128) void flash_attn(
    const u16* __restrict__ qws, const u16* __restrict__ kws,
    const u16* __restrict__ vtws, u16* __restrict__ ctxws)
{
  __shared__ __align__(16) u16 Ps[4][16 * 72];   // per-wave P scratch, padded rows
  const int qb = blockIdx.x;                     // 64-row Q block
  const int bh = blockIdx.y;
  const int b = bh >> 4, h = bh & 15;
  const int tid = threadIdx.x, lane = tid & 31, wid = tid >> 5;
  const int hf = lane >> 4, ln = lane & 15;

  const size_t headQK = (size_t)bh * Tsz * HDsz;
  const int qrow0 = qb*64 + wid*16;

  // Q fragments (A-layout): lane ln owns query row qrow0+ln, contiguous along HD
  const u16* qrow = qws + headQK + (size_t)(qrow0 + ln) * HDsz;
  v16bf qf[2];
  #pragma unroll
  for (int kk = 0; kk < 2; ++kk) qf[kk] = fragA(qrow + kk*32, hf);

  v8f o[4];
  float mrow[8], lrow[8];
  #pragma unroll
  for (int nf = 0; nf < 4; ++nf) o[nf] = vzero();
  #pragma unroll
  for (int r = 0; r < 8; ++r) { mrow[r] = -__builtin_inff(); lrow[r] = 0.f; }

  for (int kb = 0; kb <= qb; ++kb) {
    // --- S = (Q K^T) * 1/sqrt(HD), 16x64 per wave ---
    v8f s[4];
    #pragma unroll
    for (int nf = 0; nf < 4; ++nf) s[nf] = vzero();
    const u16* kbase = kws + headQK + (size_t)(kb*64) * HDsz;
    #pragma unroll
    for (int nf = 0; nf < 4; ++nf) {
      const u16* krow = kbase + (size_t)(nf*16 + ln) * HDsz;  // B-frag: lane = key
      #pragma unroll
      for (int kk = 0; kk < 2; ++kk) {
        v16bf kf = fragB(krow + kk*32, hf);
        s[nf] = wmma_bf16(qf[kk], kf, s[nf]);
      }
    }
    // scale + branchless causal mask (key > query -> -inf)
    #pragma unroll
    for (int nf = 0; nf < 4; ++nf)
      #pragma unroll
      for (int r = 0; r < 8; ++r) {
        float v = s[nf][r] * 0.125f;
        const int key = kb*64 + nf*16 + ln;
        const int qq  = qrow0 + r + 8*hf;
        s[nf][r] = (key > qq) ? -__builtin_inff() : v;
      }
    // --- online softmax: row stats live in 16-lane half-groups ---
    #pragma unroll
    for (int r = 0; r < 8; ++r) {
      float bm = fmaxf(fmaxf(s[0][r], s[1][r]), fmaxf(s[2][r], s[3][r]));
      #pragma unroll
      for (int mk = 1; mk < 16; mk <<= 1) bm = fmaxf(bm, __shfl_xor(bm, mk, 32));
      const float mnew  = fmaxf(mrow[r], bm);
      const float alpha = __expf(mrow[r] - mnew);
      mrow[r] = mnew;
      float ls = 0.f;
      #pragma unroll
      for (int nf = 0; nf < 4; ++nf) {
        float p = __expf(s[nf][r] - mnew);
        s[nf][r] = p;
        ls += p;
      }
      #pragma unroll
      for (int mk = 1; mk < 16; mk <<= 1) ls += __shfl_xor(ls, mk, 32);
      lrow[r] = lrow[r] * alpha + ls;
      #pragma unroll
      for (int nf = 0; nf < 4; ++nf) o[nf][r] *= alpha;
    }
    // --- transpose P (C-layout) to A-layout through per-wave LDS ---
    u16* pw = &Ps[wid][0];
    #pragma unroll
    for (int nf = 0; nf < 4; ++nf)
      #pragma unroll
      for (int r = 0; r < 8; ++r)
        pw[(r + 8*hf) * 72 + nf*16 + ln] = bf16bits(s[nf][r]);
    // --- O += P V  (V^T rows are contiguous along keys) ---
    const u16* vbase = vtws + (size_t)bh * HDsz * Tsz + kb*64;
    #pragma unroll
    for (int kk = 0; kk < 2; ++kk) {
      v16bf pf = fragA(pw + ln*72 + kk*32, hf);
      #pragma unroll
      for (int nf = 0; nf < 4; ++nf) {
        v16bf vf = fragB(vbase + (size_t)(nf*16 + ln) * Tsz + kk*32, hf);
        o[nf] = wmma_bf16(pf, vf, o[nf]);
      }
    }
  }

  // normalize and store ctx as bf16 in [B,T,H,HD] (= [B,T,D] row-major)
  #pragma unroll
  for (int nf = 0; nf < 4; ++nf)
    #pragma unroll
    for (int r = 0; r < 8; ++r) {
      const float v = o[nf][r] / lrow[r];
      const int tt = qrow0 + r + 8*hf;
      const int hd = nf*16 + ln;
      ctxws[(((size_t)b*Tsz + tt)*Hsz + h)*HDsz + hd] = bf16bits(v);
    }
}

// ---- kernel 3: output projection out = ctx @ Wo^T + bo ---------------------
// grid = (Mtot/128, Dsz/128), block = 256. A direct from global (bf16, contiguous K);
// Wo tile double-buffered through TDM.

__global__ __launch_bounds__(256) void out_proj(
    const u16* __restrict__ ctx, const u16* __restrict__ Wob,
    const float* __restrict__ bo, float* __restrict__ out)
{
  __shared__ __align__(16) u16 Bs[2][128 * LDST];
  const int m0 = blockIdx.x * 128, n0 = blockIdx.y * 128;
  const int tid = threadIdx.x, lane = tid & 31, wid = tid >> 5;
  const int hf = lane >> 4, ln = lane & 15;
  const int wm = wid >> 1, wn = wid & 1;

  v8f acc[2][4];
  #pragma unroll
  for (int i = 0; i < 2; ++i)
    #pragma unroll
    for (int j = 0; j < 4; ++j) acc[i][j] = vzero();

#if HAVE_TDM
  if (wid == 0)
    tdm_load_tile(Wob + (size_t)n0 * Dsz, (unsigned)(size_t)(void*)Bs[0],
                  128, KT, Dsz, Dsz);
#endif

  for (int k0 = 0, it = 0; k0 < Dsz; k0 += KT, ++it) {
    const int cur = it & 1;
#if HAVE_TDM
    const int nxt = cur ^ 1;
    if (wid == 0) {
      if (k0 + KT < Dsz) {
        tdm_load_tile(Wob + (size_t)n0 * Dsz + k0 + KT, (unsigned)(size_t)(void*)Bs[nxt],
                      128, KT, Dsz, Dsz);
        __builtin_amdgcn_s_wait_tensorcnt(1);
      } else {
        __builtin_amdgcn_s_wait_tensorcnt(0);
      }
    }
#else
    stage_copy_bf16(Bs[cur], Wob + (size_t)n0 * Dsz + k0, Dsz, tid);
#endif
    __syncthreads();
    v16bf af[2][2];
    #pragma unroll
    for (int mf = 0; mf < 2; ++mf) {
      const u16* ar = ctx + (size_t)(m0 + wm*32 + mf*16 + ln) * Dsz + k0;
      #pragma unroll
      for (int kk = 0; kk < 2; ++kk) af[mf][kk] = fragA(ar + kk*32, hf);
    }
    #pragma unroll
    for (int nf = 0; nf < 4; ++nf)
      #pragma unroll
      for (int kk = 0; kk < 2; ++kk) {
        v16bf bfr = fragB(Bs[cur] + (wn*64 + nf*16 + ln) * LDST + kk*32, hf);
        #pragma unroll
        for (int mf = 0; mf < 2; ++mf)
          acc[mf][nf] = wmma_bf16(af[mf][kk], bfr, acc[mf][nf]);
      }
    __syncthreads();
  }

  #pragma unroll
  for (int nf = 0; nf < 4; ++nf) {
    const float bias = bo[n0 + wn*64 + nf*16 + ln];
    #pragma unroll
    for (int mf = 0; mf < 2; ++mf)
      #pragma unroll
      for (int r = 0; r < 8; ++r) {
        const int m = m0 + wm*32 + mf*16 + r + 8*hf;
        const int n = n0 + wn*64 + nf*16 + ln;
        out[(size_t)m * Dsz + n] = acc[mf][nf][r] + bias;
      }
  }
}

// ---- host entry ------------------------------------------------------------

extern "C" void kernel_launch(void* const* d_in, const int* in_sizes, int n_in,
                              void* d_out, int out_size, void* d_ws, size_t ws_size,
                              hipStream_t stream)
{
  (void)in_sizes; (void)n_in; (void)out_size; (void)ws_size;
  const float* x  = (const float*)d_in[0];
  const float* Wq = (const float*)d_in[1];
  const float* Wk = (const float*)d_in[2];
  const float* Wv = (const float*)d_in[3];
  const float* Wo = (const float*)d_in[4];
  const float* bo = (const float*)d_in[5];

  // Workspace (bf16): xb | Wq | Wk | Wv | Wo | Q | K | V^T | ctx  (~48 MB)
  const size_t XE = (size_t)Mtot * Dsz;            // 4M elements
  const size_t DD = (size_t)Dsz * Dsz;             // 1M elements
  const size_t HE = (size_t)Bsz * Hsz * Tsz * HDsz;// 4M elements
  u16* xb    = (u16*)d_ws;
  u16* wqb   = xb   + XE;
  u16* wkb   = wqb  + DD;
  u16* wvb   = wkb  + DD;
  u16* wob   = wvb  + DD;
  u16* qws   = wob  + DD;
  u16* kws   = qws  + HE;
  u16* vtws  = kws  + HE;
  u16* ctxws = vtws + HE;

  // prepack fp32 -> bf16 (one pass each)
  cvt_bf16<<<dim3((unsigned)(XE/4/256)), 256, 0, stream>>>(x,  xb,  (int)(XE/4));
  cvt_bf16<<<dim3((unsigned)(DD/4/256)), 256, 0, stream>>>(Wq, wqb, (int)(DD/4));
  cvt_bf16<<<dim3((unsigned)(DD/4/256)), 256, 0, stream>>>(Wk, wkb, (int)(DD/4));
  cvt_bf16<<<dim3((unsigned)(DD/4/256)), 256, 0, stream>>>(Wv, wvb, (int)(DD/4));
  cvt_bf16<<<dim3((unsigned)(DD/4/256)), 256, 0, stream>>>(Wo, wob, (int)(DD/4));

  qkv_wmma  <<<dim3(Mtot/128, Dsz/128, 3), 256, 0, stream>>>(xb, wqb, wkb, wvb, qws, kws, vtws);
  flash_attn<<<dim3(Tsz/64, Bsz*Hsz),      128, 0, stream>>>(qws, kws, vtws, ctxws);
  out_proj  <<<dim3(Mtot/128, Dsz/128),    256, 0, stream>>>(ctxws, wob, bo, (float*)d_out);
}
